// SelfAttentionBlock_15839839388316
// MI455X (gfx1250) — compile-verified, matrix operand order checked by reading, not run on hardware
//
#include <hip/hip_runtime.h>
#include <hip/hip_bf16.h>

// Problem constants (match reference: B,S,E,D = 4,2048,768,64)
constexpr int Bn = 4;
constexpr int Sn = 2048;
constexpr int En = 768;
constexpr int Dn = 64;

typedef __attribute__((ext_vector_type(16))) __bf16 v16bf;
typedef __attribute__((ext_vector_type(8)))  float  v8f;
typedef __attribute__((ext_vector_type(4)))  unsigned int v4u;
typedef __attribute__((ext_vector_type(8)))  int    v8i;
typedef __attribute__((ext_vector_type(4)))  int    v4i;

#if defined(__has_builtin)
#if __has_builtin(__builtin_amdgcn_tensor_load_to_lds) && \
    __has_builtin(__builtin_amdgcn_s_wait_tensorcnt)
#define USE_TDM 1
#endif
#endif
#ifndef USE_TDM
#define USE_TDM 0
#endif

union Frag16 {
    v16bf v;
    uint4 q[2];
    unsigned short h[16];
};

// RNE fp32 -> bf16 (bits in low 16 of high half)
__device__ __forceinline__ unsigned rne_hi(float f) {
    union { float f; unsigned u; } v; v.f = f;
    return v.u + 0x7fffu + ((v.u >> 16) & 1u);
}
// Pack two fp32 into {bf16(hi), bf16(lo)} with one v_perm_b32.
__device__ __forceinline__ unsigned pack_bf16x2(float lo, float hi) {
    return __builtin_amdgcn_perm(rne_hi(hi), rne_hi(lo), 0x07060302u);
}
__device__ __forceinline__ unsigned short f2bfu(float f) {
    return (unsigned short)(rne_hi(f) >> 16);
}

__device__ __forceinline__ v8f wmma_bf16(v16bf a, v16bf b, v8f c) {
    return __builtin_amdgcn_wmma_f32_16x16x32_bf16(
        /*neg_a=*/false, a, /*neg_b=*/false, b,
        /*c_mod=*/(short)0, c, /*reuse_a=*/false, /*reuse_b=*/false);
}

// A/B fragment (16x32 / 32x16 bf16) from a source contiguous along K.
// Layout (ISA 7.12.2): lane<16 -> K = {k0..k0+7, k0+16..k0+23}, lane>=16 -> +8.
// Each half of the fragment is 8 contiguous bf16 = one b128 load.
__device__ __forceinline__ v16bf load_frag(const unsigned short* rowp,
                                           int lanehi, int k0) {
    Frag16 f;
    int ks = k0 + lanehi * 8;
    f.q[0] = *(const uint4*)(rowp + ks);
    f.q[1] = *(const uint4*)(rowp + ks + 16);
    return f.v;
}

__device__ __forceinline__ float redmax16(float v) {
    v = fmaxf(v, __shfl_xor(v, 1, 16));
    v = fmaxf(v, __shfl_xor(v, 2, 16));
    v = fmaxf(v, __shfl_xor(v, 4, 16));
    v = fmaxf(v, __shfl_xor(v, 8, 16));
    return v;
}
__device__ __forceinline__ float redsum16(float v) {
    v += __shfl_xor(v, 1, 16);
    v += __shfl_xor(v, 2, 16);
    v += __shfl_xor(v, 4, 16);
    v += __shfl_xor(v, 8, 16);
    return v;
}

#if USE_TDM
// Issue one TDM 2-D tile load: global (row-major, 2-byte elems, row stride
// stride0 elems) -> LDS (rows packed contiguously). D# per ISA §8.3/8.4.
__device__ __forceinline__ void tdm_load_2d(unsigned lds_addr, const void* gptr,
                                            unsigned dim0, unsigned dim1,
                                            unsigned stride0,
                                            unsigned tile0, unsigned tile1) {
    unsigned long long ga = (unsigned long long)(size_t)gptr;
    v4u g0;
    g0.x = 1u;                                            // count=1, user mode
    g0.y = lds_addr;                                      // LDS byte address
    g0.z = (unsigned)(ga & 0xffffffffull);                // global_addr[31:0]
    g0.w = (unsigned)((ga >> 32) & 0x01ffffffull) | (2u << 30); // [56:32]|type=2
    v8i g1;
    g1[0] = 0x00010000;                                   // data_size=1 (2B)
    g1[1] = (int)((dim0 & 0xffffu) << 16);                // tensor_dim0[15:0]
    g1[2] = (int)(((dim0 >> 16) & 0xffffu) | ((dim1 & 0xffffu) << 16));
    g1[3] = (int)(((dim1 >> 16) & 0xffffu) | ((tile0 & 0xffffu) << 16));
    g1[4] = (int)(tile1 & 0xffffu);                       // tile_dim1 (tile2=0)
    g1[5] = (int)stride0;                                 // dim0_stride[31:0]
    g1[6] = 0;
    g1[7] = 0;
    v4i gz = {};
#if __clang_major__ >= 23
    v8i gz8 = {};
    __builtin_amdgcn_tensor_load_to_lds(g0, g1, gz, gz, gz8, 0);
#else
    __builtin_amdgcn_tensor_load_to_lds(g0, g1, gz, gz, 0);
#endif
}
#endif

// ---------------------------------------------------------------------------
// Prep 1: x fp32 [B,S,E] -> bf16 [B,S,E]. 8 floats per thread, b128 in/out.
// ---------------------------------------------------------------------------
__global__ __launch_bounds__(256)
void cvt_x_kernel(const float* __restrict__ x, unsigned* __restrict__ xbf) {
    const size_t i = (size_t)blockIdx.x * blockDim.x + threadIdx.x; // 8 floats
    const float4* p = (const float4*)x + i * 2;
    const float4 f0 = p[0];
    const float4 f1 = p[1];
    uint4 o;
    o.x = pack_bf16x2(f0.x, f0.y);
    o.y = pack_bf16x2(f0.z, f0.w);
    o.z = pack_bf16x2(f1.x, f1.y);
    o.w = pack_bf16x2(f1.z, f1.w);
    ((uint4*)xbf)[i] = o;
}

// ---------------------------------------------------------------------------
// Prep 2: W fp32 [E,D] -> bf16 transposed [D,E] (contiguous along E).
// ---------------------------------------------------------------------------
__global__ __launch_bounds__(256)
void cvt_wt_kernel(const float* __restrict__ W, unsigned* __restrict__ Wt) {
    const int i = blockIdx.x * blockDim.x + threadIdx.x;   // D*E/2 threads
    const int d = i / (En / 2);
    const int e = (i % (En / 2)) * 2;
    const float lo = W[(size_t)e * Dn + d];
    const float hi = W[(size_t)(e + 1) * Dn + d];
    Wt[(size_t)d * (En / 2) + (e >> 1)] = pack_bf16x2(lo, hi);
}

// ---------------------------------------------------------------------------
// Kernel 1: QKV projection. One wave -> one 16(row) x 64(col) tile of Q,K,V.
// Pure b128 fragment loads (x bf16 row-major, W bf16 transposed) + WMMA.
// Q,K stored bf16 [B,S,D]; V stored bf16 transposed [B,D,S].
// ---------------------------------------------------------------------------
__global__ __launch_bounds__(128)
void qkv_proj_kernel(const unsigned short* __restrict__ xbf,
                     const unsigned short* __restrict__ Wtq,
                     const unsigned short* __restrict__ Wtk,
                     const unsigned short* __restrict__ Wtv,
                     const float* __restrict__ bq,
                     const float* __restrict__ bk,
                     const float* __restrict__ bv,
                     unsigned short* __restrict__ Qo,
                     unsigned short* __restrict__ Ko,
                     unsigned short* __restrict__ Vt) {
    const int lane   = threadIdx.x & 31;
    const int lanelo = lane & 15;
    const int lanehi = lane >> 4;
    const int wave   = blockIdx.x * (blockDim.x >> 5) + (threadIdx.x >> 5);

    const int tilesPerB = Sn / 16;
    const int b    = wave / tilesPerB;
    const int srow = (wave % tilesPerB) * 16;

    const unsigned short* xrow = xbf + (size_t)(b * Sn + srow + lanelo) * En;

    v8f accq[4] = {}, acck[4] = {}, accv[4] = {};

    for (int e0 = 0; e0 < En; e0 += 32) {
        const v16bf a = load_frag(xrow, lanehi, e0);
#pragma unroll
        for (int n = 0; n < 4; ++n) {
            const size_t woff = (size_t)(n * 16 + lanelo) * En;
            accq[n] = wmma_bf16(a, load_frag(Wtq + woff, lanehi, e0), accq[n]);
            acck[n] = wmma_bf16(a, load_frag(Wtk + woff, lanehi, e0), acck[n]);
            accv[n] = wmma_bf16(a, load_frag(Wtv + woff, lanehi, e0), accv[n]);
        }
        if (e0 + 32 < En) __builtin_prefetch(xrow + e0 + 32, 0, 3);
    }

#pragma unroll
    for (int n = 0; n < 4; ++n) {
        const int col = n * 16 + lanelo;
        const float aq = bq[col], ak = bk[col], av = bv[col];
#pragma unroll
        for (int g = 0; g < 8; ++g) {
            const int r = srow + g + lanehi * 8;
            const size_t idx = ((size_t)b * Sn + r) * Dn + col;
            Qo[idx] = f2bfu(accq[n][g] + aq);
            Ko[idx] = f2bfu(acck[n][g] + ak);
            Vt[((size_t)b * Dn + col) * Sn + r] = f2bfu(accv[n][g] + av);
        }
    }
}

// ---------------------------------------------------------------------------
// Kernel 2: flash attention. 4 waves/block share one batch's K/V stream.
// K/V tiles (32 keys) staged into LDS by the Tensor Data Mover, double-
// buffered: wave 0 issues next-stage tensor_load_to_lds at the top of the
// iteration; s_wait_tensorcnt + barrier at the bottom. All fragment loads in
// the hot loop are ds_load_b128. Online softmax, P through LDS (packed b32).
// ---------------------------------------------------------------------------
constexpr int KT_OFF = 0;                 // 2 x 32*64 bf16 (2 x 4KB)
constexpr int VT_OFF = 2 * 32 * Dn;       // 2 x 64*32 bf16 (2 x 4KB)
constexpr int P_OFF  = VT_OFF + 2 * Dn * 32;  // 4 x 16*32 bf16 (4 x 1KB)
constexpr int SMEM_ELTS = P_OFF + 4 * 16 * 32;  // 10240 ushorts = 20KB

__global__ __launch_bounds__(128)
void attn_kernel(const unsigned short* __restrict__ Q,
                 const unsigned short* __restrict__ K,
                 const unsigned short* __restrict__ Vt,
                 float* __restrict__ out) {
    __shared__ __attribute__((aligned(16))) unsigned short smem[SMEM_ELTS];

    const int wv     = threadIdx.x >> 5;
    const int lane   = threadIdx.x & 31;
    const int lanelo = lane & 15;
    const int lanehi = lane >> 4;
    const int wave   = blockIdx.x * 4 + wv;

    const int tilesPerB = Sn / 16;
    const int b    = wave / tilesPerB;   // same for all 4 waves in a block
    const int qrow = (wave % tilesPerB) * 16;

    unsigned short* pbuf  = smem + P_OFF + wv * (16 * 32);
    unsigned*       pbuf32 = (unsigned*)pbuf;

    const unsigned short* Qb = Q  + (size_t)b * Sn * Dn;
    const unsigned short* Kb = K  + (size_t)b * Sn * Dn;
    const unsigned short* Vb = Vt + (size_t)b * Dn * Sn;

    // Q fragments for d = 0..31 and 32..63 (held for the whole key loop).
    const unsigned short* qptr = Qb + (size_t)(qrow + lanelo) * Dn;
    const v16bf aq0 = load_frag(qptr, lanehi, 0);
    const v16bf aq1 = load_frag(qptr, lanehi, 32);

#if USE_TDM
    const unsigned ldsbase = (unsigned)(size_t)&smem[0];
    // Prologue: stage tiles for kb=0 into buffer 0.
    if (wv == 0) {
        tdm_load_2d(ldsbase + KT_OFF * 2, Kb, Dn, Sn, Dn, Dn, 32);
        tdm_load_2d(ldsbase + VT_OFF * 2, Vb, Sn, Dn, Sn, 32, Dn);
    }
    __builtin_amdgcn_s_wait_tensorcnt(0);
    __syncthreads();
#endif

    v8f acc[4] = {};
    float row_m[8], row_l[8];
#pragma unroll
    for (int g = 0; g < 8; ++g) { row_m[g] = -1e30f; row_l[g] = 0.0f; }

    const float LOG2E = 1.4426950408889634f;

    for (int kb = 0; kb < Sn; kb += 32) {
        const int buf = (kb >> 5) & 1;
        const unsigned short* kt = smem + KT_OFF + buf * (32 * Dn);
        const unsigned short* vt = smem + VT_OFF + buf * (Dn * 32);

#if USE_TDM
        // Issue next stage into the other buffer (overlaps this iteration).
        if (wv == 0 && kb + 32 < Sn) {
            const int nb = buf ^ 1;
            tdm_load_2d(ldsbase + (KT_OFF + nb * 32 * Dn) * 2,
                        Kb + (size_t)(kb + 32) * Dn, Dn, Sn, Dn, Dn, 32);
            tdm_load_2d(ldsbase + (VT_OFF + nb * Dn * 32) * 2,
                        Vb + (kb + 32), Sn, Dn, Sn, 32, Dn);
        }
#else
        // Fallback: cooperative global->LDS staging of the current tile.
        {
            const int t = threadIdx.x;
            const uint4* gk = (const uint4*)(Kb + (size_t)kb * Dn);
            uint4* lk = (uint4*)(smem + KT_OFF + buf * (32 * Dn));
            lk[t] = gk[t];
            lk[t + 128] = gk[t + 128];
            const unsigned short* vsrc = Vb + kb;
            const int d = t >> 1, part = t & 1;
            uint4* lv = (uint4*)(smem + VT_OFF + buf * (Dn * 32));
            lv[d * 4 + part * 2]     = *(const uint4*)(vsrc + (size_t)d * Sn + part * 16);
            lv[d * 4 + part * 2 + 1] = *(const uint4*)(vsrc + (size_t)d * Sn + part * 16 + 8);
            __syncthreads();
        }
#endif

        // Scores from LDS. tile0: even keys kb+2n ; tile1: odd keys kb+2n+1
        const unsigned short* k0p = kt + (size_t)(2 * lanelo) * Dn;
        const unsigned short* k1p = k0p + Dn;
        v8f s0 = {}, s1 = {};
        s0 = wmma_bf16(aq0, load_frag(k0p, lanehi, 0),  s0);
        s0 = wmma_bf16(aq1, load_frag(k0p, lanehi, 32), s0);
        s1 = wmma_bf16(aq0, load_frag(k1p, lanehi, 0),  s1);
        s1 = wmma_bf16(aq1, load_frag(k1p, lanehi, 32), s1);

        // Online softmax update per row.
#pragma unroll
        for (int g = 0; g < 8; ++g) {
            const float a = s0[g], c = s1[g];
            const float mx   = redmax16(fmaxf(a, c));
            const float mnew = fmaxf(row_m[g], mx);
            const float scale = __builtin_amdgcn_exp2f((row_m[g] - mnew) * LOG2E);
            const float p0 = __builtin_amdgcn_exp2f((a - mnew) * LOG2E);
            const float p1 = __builtin_amdgcn_exp2f((c - mnew) * LOG2E);
            row_l[g] = row_l[g] * scale + redsum16(p0 + p1);
            row_m[g] = mnew;
            acc[0][g] *= scale; acc[1][g] *= scale;
            acc[2][g] *= scale; acc[3][g] *= scale;
            const int r = g + lanehi * 8;
            pbuf32[r * 16 + lanelo] = pack_bf16x2(p0, p1);  // keys 2n, 2n+1
        }

        __syncthreads();                                    // P visible in LDS
        const v16bf pa = load_frag(pbuf + lanelo * 32, lanehi, 0);

        // O += P V  (V tile in LDS, keys contiguous).
#pragma unroll
        for (int n = 0; n < 4; ++n) {
            const unsigned short* vp = vt + (size_t)(n * 16 + lanelo) * 32;
            acc[n] = wmma_bf16(pa, load_frag(vp, lanehi, 0), acc[n]);
        }

#if USE_TDM
        __builtin_amdgcn_s_wait_tensorcnt(0);  // next tiles landed (wave 0)
#endif
        __syncthreads();  // P reusable + staged tiles visible to all waves
    }

    float* op = out + ((size_t)b * Sn + qrow) * Dn;
#pragma unroll
    for (int n = 0; n < 4; ++n) {
#pragma unroll
        for (int g = 0; g < 8; ++g) {
            const int r = g + lanehi * 8;
            op[(size_t)r * Dn + n * 16 + lanelo] = acc[n][g] / row_l[g];
        }
    }
}

extern "C" void kernel_launch(void* const* d_in, const int* in_sizes, int n_in,
                              void* d_out, int out_size, void* d_ws, size_t ws_size,
                              hipStream_t stream) {
    const float* x  = (const float*)d_in[0];
    const float* Wq = (const float*)d_in[1];
    const float* bq = (const float*)d_in[2];
    const float* Wk = (const float*)d_in[3];
    const float* bk = (const float*)d_in[4];
    const float* Wv = (const float*)d_in[5];
    const float* bv = (const float*)d_in[6];

    // Workspace layout (bf16 halves):
    unsigned short* xbf = (unsigned short*)d_ws;              // [B,S,E]
    unsigned short* Wtq = xbf + (size_t)Bn * Sn * En;         // [D,E]
    unsigned short* Wtk = Wtq + (size_t)Dn * En;              // [D,E]
    unsigned short* Wtv = Wtk + (size_t)Dn * En;              // [D,E]
    unsigned short* Qw  = Wtv + (size_t)Dn * En;              // [B,S,D]
    unsigned short* Kw  = Qw  + (size_t)Bn * Sn * Dn;         // [B,S,D]
    unsigned short* Vt  = Kw  + (size_t)Bn * Sn * Dn;         // [B,D,S]
    // total ~16 MB

    // Prep conversions
    {
        const int nthr = (Bn * Sn * En) / 8;                  // 786432
        cvt_x_kernel<<<nthr / 256, 256, 0, stream>>>(x, (unsigned*)xbf);
        const int wthr = (Dn * En) / 2;                       // 24576
        cvt_wt_kernel<<<wthr / 256, 256, 0, stream>>>(Wq, (unsigned*)Wtq);
        cvt_wt_kernel<<<wthr / 256, 256, 0, stream>>>(Wk, (unsigned*)Wtk);
        cvt_wt_kernel<<<wthr / 256, 256, 0, stream>>>(Wv, (unsigned*)Wtv);
    }

    const int tiles = (Bn * Sn) / 16;     // 512 waves
    const dim3 block(128);                // 4 waves per block (wave32)
    const dim3 grid(tiles / 4);           // 128 blocks

    qkv_proj_kernel<<<grid, block, 0, stream>>>(xbf, Wtq, Wtk, Wtv,
                                                bq, bk, bv, Qw, Kw, Vt);
    attn_kernel<<<grid, block, 0, stream>>>(Qw, Kw, Vt, (float*)d_out);
}
